// ALU_29480655519972
// MI455X (gfx1250) — matrix operand order repaired and checked by reading, not compiled, not run on hardware
//
#include <hip/hip_runtime.h>
#include <cstdint>
#include <cstddef>

// ---------------- problem dims (from reference) ----------------
#define H_NUM   16
#define B_DIM   2
#define N_DIM   2048
#define E_DIM   2048
#define QK_DIM  2048
#define M_DIM   4096
#define V_DIM   2048
#define DK      (QK_DIM / H_NUM)   // 128
#define DV      (V_DIM / H_NUM)    // 128
#define BN_ROWS (B_DIM * N_DIM)    // 4096

typedef __attribute__((ext_vector_type(8)))  float   v8f;
typedef __attribute__((ext_vector_type(16))) __bf16  v16bf;
typedef __attribute__((ext_vector_type(8)))  __bf16  v8bf;
typedef __attribute__((ext_vector_type(4)))  __bf16  v4bf;

union ABFrag { v16bf v; v8bf h[2]; };

// silu via hardware reciprocal: v_exp_f32 + v_rcp_f32 + mul.
// (bf16-rounded output => correctly-rounded IEEE divide is wasted work)
__device__ __forceinline__ float silu_f(float x) {
  const float s = __builtin_amdgcn_rcpf(1.0f + __expf(-x));
  return x * s;
}

__device__ __forceinline__ v8f wmma_bf16(v16bf a, v16bf b, v8f c) {
  // v_wmma_f32_16x16x32_bf16  D = A(16x32) * B(32x16) + C(16x16 f32)
  return __builtin_amdgcn_wmma_f32_16x16x32_bf16(
      /*neg_a=*/false, a, /*neg_b=*/false, b,
      /*c_mod=*/(short)0, c, /*reuse_a=*/false, /*reuse_b=*/false);
}

// A fragment (16 rows x 32 K, bf16). ISA layout: lane<16 -> row=lane,
// K = {0..7, 16..23}; lane>=16 -> row=lane-16, K = {8..15, 24..31}.
// Both halves are contiguous 8-element runs -> two 16B LDS loads.
__device__ __forceinline__ v16bf load_a_frag(const __bf16* lds, int row0,
                                             int stride, int k0, int lane) {
  const int half = lane >> 4;
  const int r    = lane & 15;
  const __bf16* p = lds + (row0 + r) * stride + k0 + half * 8;
  ABFrag f;
  f.h[0] = *(const v8bf*)p;
  f.h[1] = *(const v8bf*)(p + 16);
  return f.v;
}

// B fragment (32 K x 16 cols, bf16), same K layout as A: lane<16 ->
// col=lane, K={0..7,16..23}; lane>=16 -> col=lane-16, K={8..15,24..31}.
__device__ __forceinline__ v16bf load_b_frag(const __bf16* lds, int col0,
                                             int stride, int k0, int lane) {
  const int kh = lane >> 4;
  const int c  = lane & 15;
  const __bf16* p = lds + (col0 + c) * stride + k0 + kh * 8;
  ABFrag f;
  f.h[0] = *(const v8bf*)p;
  f.h[1] = *(const v8bf*)(p + 16);
  return f.v;
}

// ---------------- f32 -> bf16 convert (optionally scaled) ----------------
__global__ __launch_bounds__(256)
void cvt_kernel(const float* __restrict__ s, __bf16* __restrict__ d, float sc) {
  const size_t i = ((size_t)blockIdx.x * blockDim.x + threadIdx.x) * 4;
  float4 f = *(const float4*)(s + i);
  v4bf o;
  o[0] = (__bf16)(f.x * sc);
  o[1] = (__bf16)(f.y * sc);
  o[2] = (__bf16)(f.z * sc);
  o[3] = (__bf16)(f.w * sc);
  *(v4bf*)(d + i) = o;
}

// ---------------- generic bf16 GEMM: C = A[M,K] * Bt[N,K]^T ----------------
// mode 0: store bf16, mode 1: store bf16(silu(x)), mode 2: store f32
__global__ __launch_bounds__(256)
void gemm_bt_kernel(const __bf16* __restrict__ A, const __bf16* __restrict__ Bt,
                    __bf16* __restrict__ outb, float* __restrict__ outf,
                    int Ncols, int Kdim, int mode) {
  constexpr int LDT = 40;  // 32 + 8 pad (keeps 16B alignment, spreads banks)
  __shared__ __bf16 As[128 * LDT];
  __shared__ __bf16 Bs[128 * LDT];

  const int tid  = threadIdx.x;
  const int lane = tid & 31;
  const int w    = tid >> 5;      // 8 waves (wave32)
  const int wm   = w >> 1;        // 0..3 -> 32-row strip
  const int wn   = w & 1;         // 0..1 -> 64-col strip
  const long blockM = (long)blockIdx.y * 128;
  const long blockN = (long)blockIdx.x * 128;

  v8f acc[2][4];
#pragma unroll
  for (int i = 0; i < 2; ++i)
#pragma unroll
    for (int j = 0; j < 4; ++j) acc[i][j] = (v8f)0.0f;

  for (int kt = 0; kt < Kdim; kt += 32) {
#pragma unroll
    for (int i = 0; i < 2; ++i) {  // 2 x 256 threads x 8 elems = 4096 per tile
      const int e = (tid + i * 256) * 8;
      const int r = e >> 5, c = e & 31;
      *(uint4*)&As[r * LDT + c] =
          *(const uint4*)&A[(blockM + r) * (long)Kdim + kt + c];
      *(uint4*)&Bs[r * LDT + c] =
          *(const uint4*)&Bt[(blockN + r) * (long)Kdim + kt + c];
    }
    __syncthreads();

    v16bf af[2];
#pragma unroll
    for (int mf = 0; mf < 2; ++mf)
      af[mf] = load_a_frag(As, wm * 32 + mf * 16, LDT, 0, lane);
#pragma unroll
    for (int nf = 0; nf < 4; ++nf) {
      v16bf bfr = load_b_frag(Bs, wn * 64 + nf * 16, LDT, 0, lane);
#pragma unroll
      for (int mf = 0; mf < 2; ++mf)
        acc[mf][nf] = wmma_bf16(af[mf], bfr, acc[mf][nf]);
    }
    __syncthreads();
  }

  const int r0 = (lane >> 4) * 8;
  const int cc = lane & 15;
#pragma unroll
  for (int mf = 0; mf < 2; ++mf)
#pragma unroll
    for (int nf = 0; nf < 4; ++nf)
#pragma unroll
      for (int i = 0; i < 8; ++i) {
        const long row = blockM + wm * 32 + mf * 16 + r0 + i;
        const long col = blockN + wn * 64 + nf * 16 + cc;
        float vvv = acc[mf][nf][i];
        if (mode == 1) vvv = silu_f(vvv);
        if (mode == 2) outf[row * Ncols + col] = vvv;
        else           outb[row * Ncols + col] = (__bf16)vvv;
      }
}

// ---------------- fused attention: out(b,h) = silu(q k^T) v, gated ----------------
// grid: (N/128, B*H). Per block: 128 query rows, all dv=128 cols of one head.
__global__ __launch_bounds__(256)
void attn_kernel(const __bf16* __restrict__ qb, const __bf16* __restrict__ kb,
                 const __bf16* __restrict__ vb, const __bf16* __restrict__ gateb,
                 __bf16* __restrict__ gatedb) {
  constexpr int LDQ = 136;  // 128 + 8 pad
  constexpr int LDE = 40;   // 32 + 8 pad
  __shared__ __bf16 Qs[128 * LDQ];  // q tile   [n=128][d=128]
  __shared__ __bf16 Ks[32 * LDQ];   // k chunk  [m=32][d=128]
  __shared__ __bf16 Vs[128 * LDE];  // v chunk transposed [d=128][m=32]
  __shared__ __bf16 Es[128 * LDE];  // silu(energy) chunk [n=128][m=32]

  const int tid  = threadIdx.x;
  const int lane = tid & 31;
  const int w    = tid >> 5;
  const int wm   = w >> 1, wn = w & 1;
  const int bh   = blockIdx.y;
  const int b    = bh / H_NUM, h = bh % H_NUM;
  const long rowBase = (long)b * N_DIM + (long)blockIdx.x * 128;

  // stage Q tile once: 16384 bf16 -> 8 x (256 thr x 16B)
#pragma unroll
  for (int i = 0; i < 8; ++i) {
    const int e = (tid + i * 256) * 8;
    const int r = e >> 7, d = e & 127;
    *(uint4*)&Qs[r * LDQ + d] =
        *(const uint4*)&qb[(rowBase + r) * QK_DIM + h * DK + d];
  }

  v8f oacc[2][4];
#pragma unroll
  for (int i = 0; i < 2; ++i)
#pragma unroll
    for (int j = 0; j < 4; ++j) oacc[i][j] = (v8f)0.0f;

  for (int mc = 0; mc < M_DIM; mc += 32) {
    __syncthreads();  // previous iter's Es/Vs reads done; Qs ready (iter 0)
    // k chunk [32 x 128]
#pragma unroll
    for (int i = 0; i < 2; ++i) {
      const int e = (tid + i * 256) * 8;
      const int r = e >> 7, d = e & 127;
      *(uint4*)&Ks[r * LDQ + d] =
          *(const uint4*)&kb[(long)(mc + r) * QK_DIM + h * DK + d];
    }
    // v chunk, transposed into Vs[d][m]
#pragma unroll
    for (int g = 0; g < 2; ++g) {
      const int idx = tid + g * 256;
      const int m = idx & 31, d0 = (idx >> 5) * 8;
      v8bf t = *(const v8bf*)&vb[(long)(mc + m) * V_DIM + h * DV + d0];
#pragma unroll
      for (int j = 0; j < 8; ++j) Vs[(d0 + j) * LDE + m] = t[j];
    }
    __syncthreads();

    // phase 1: E[128x32] = Q[128x128] * Kchunk^T ; wave w owns rows 16w..16w+15
    v8f e0 = (v8f)0.0f, e1 = (v8f)0.0f;
#pragma unroll
    for (int kt = 0; kt < 128; kt += 32) {
      v16bf af = load_a_frag(Qs, w * 16, LDQ, kt, lane);
      v16bf b0 = load_b_frag(Ks, 0,  LDQ, kt, lane);
      v16bf b1 = load_b_frag(Ks, 16, LDQ, kt, lane);
      e0 = wmma_bf16(af, b0, e0);
      e1 = wmma_bf16(af, b1, e1);
    }
    {
      const int r0 = (lane >> 4) * 8, cc = lane & 15;
#pragma unroll
      for (int i = 0; i < 8; ++i) {
        Es[(w * 16 + r0 + i) * LDE + cc]      = (__bf16)silu_f(e0[i]);
        Es[(w * 16 + r0 + i) * LDE + 16 + cc] = (__bf16)silu_f(e1[i]);
      }
    }
    __syncthreads();

    // phase 2: oacc += E[128x32] * Vchunk[32x128]; wave tile 32x64
    v16bf af2[2];
#pragma unroll
    for (int mf = 0; mf < 2; ++mf)
      af2[mf] = load_a_frag(Es, wm * 32 + mf * 16, LDE, 0, lane);
#pragma unroll
    for (int nf = 0; nf < 4; ++nf) {
      v16bf bfr = load_b_frag(Vs, wn * 64 + nf * 16, LDE, 0, lane);
#pragma unroll
      for (int mf = 0; mf < 2; ++mf)
        oacc[mf][nf] = wmma_bf16(af2[mf], bfr, oacc[mf][nf]);
    }
  }

  // epilogue: multiply by precomputed silu-gate, write bf16 for final GEMM
  const int r0 = (lane >> 4) * 8, cc = lane & 15;
#pragma unroll
  for (int mf = 0; mf < 2; ++mf)
#pragma unroll
    for (int nf = 0; nf < 4; ++nf)
#pragma unroll
      for (int i = 0; i < 8; ++i) {
        const long row = rowBase + wm * 32 + mf * 16 + r0 + i;
        const int  col = h * DV + wn * 64 + nf * 16 + cc;
        const long idx = row * (long)V_DIM + col;
        const float g = (float)gateb[idx];
        gatedb[idx] = (__bf16)(oacc[mf][nf][i] * g);
      }
}

// ---------------- host side ----------------
extern "C" void kernel_launch(void* const* d_in, const int* in_sizes, int n_in,
                              void* d_out, int out_size, void* d_ws, size_t ws_size,
                              hipStream_t stream) {
  (void)in_sizes; (void)n_in; (void)out_size; (void)ws_size;
  const float* x  = (const float*)d_in[0];  // [B,N,E]
  const float* Wq = (const float*)d_in[1];  // [QK,E]
  const float* kw = (const float*)d_in[2];  // [M,QK]
  const float* vw = (const float*)d_in[3];  // [M,V]
  const float* Wg = (const float*)d_in[4];  // [V,E]
  const float* Wo = (const float*)d_in[5];  // [E,V]
  float* out = (float*)d_out;               // [B,N,E] f32

  __bf16* ws = (__bf16*)d_ws;
  size_t off = 0;
  __bf16* xb     = ws + off; off += (size_t)BN_ROWS * E_DIM;
  __bf16* Wqb    = ws + off; off += (size_t)QK_DIM * E_DIM;
  __bf16* Wgb    = ws + off; off += (size_t)V_DIM * E_DIM;
  __bf16* Wob    = ws + off; off += (size_t)E_DIM * V_DIM;
  __bf16* kb     = ws + off; off += (size_t)M_DIM * QK_DIM;
  __bf16* vb     = ws + off; off += (size_t)M_DIM * V_DIM;
  __bf16* qb     = ws + off; off += (size_t)BN_ROWS * QK_DIM;
  __bf16* gateb  = ws + off; off += (size_t)BN_ROWS * V_DIM;
  __bf16* gatedb = ws + off; off += (size_t)BN_ROWS * V_DIM;

  const float scale = 11.313708498984761f;  // sqrt(dk) = sqrt(128)

  auto cvt = [&](const float* s, __bf16* d, size_t n, float sc) {
    cvt_kernel<<<(unsigned)(n / 1024), 256, 0, stream>>>(s, d, sc);
  };
  cvt(x,  xb,  (size_t)BN_ROWS * E_DIM, 1.0f);
  cvt(Wq, Wqb, (size_t)QK_DIM * E_DIM, 1.0f);
  cvt(Wg, Wgb, (size_t)V_DIM * E_DIM, 1.0f);
  cvt(Wo, Wob, (size_t)E_DIM * V_DIM, 1.0f);
  cvt(kw, kb,  (size_t)M_DIM * QK_DIM, scale);
  cvt(vw, vb,  (size_t)M_DIM * V_DIM, scale);

  // q = x * Wq^T  -> bf16 [BN, QK]
  gemm_bt_kernel<<<dim3(QK_DIM / 128, BN_ROWS / 128), 256, 0, stream>>>(
      xb, Wqb, qb, nullptr, QK_DIM, E_DIM, /*mode=*/0);

  // gate = silu(x * Wg^T) -> bf16 [BN, V]
  gemm_bt_kernel<<<dim3(V_DIM / 128, BN_ROWS / 128), 256, 0, stream>>>(
      xb, Wgb, gateb, nullptr, V_DIM, E_DIM, /*mode=*/1);

  // fused attention + gating -> bf16 [BN, V]
  attn_kernel<<<dim3(N_DIM / 128, B_DIM * H_NUM), 256, 0, stream>>>(
      qb, kb, vb, gateb, gatedb);

  // final = gated * Wout^T -> f32 d_out [BN, E]
  gemm_bt_kernel<<<dim3(E_DIM / 128, BN_ROWS / 128), 256, 0, stream>>>(
      gatedb, Wob, nullptr, out, E_DIM, V_DIM, /*mode=*/2);
}
